// Architecture_43559558316192
// MI455X (gfx1250) — matrix operand order repaired and checked by reading, not compiled, not run on hardware
//
#include <hip/hip_runtime.h>

// ---------------------------------------------------------------------------
// CDNA5 (gfx1250) AKT transformer: bf16 WMMA GEMMs (TDM-staged through LDS)
// + fused AKT attention. wave32; V_WMMA_F32_16X16X32_BF16 for matrix math.
// ---------------------------------------------------------------------------

typedef __attribute__((ext_vector_type(16))) __bf16        v16bf;
typedef __attribute__((ext_vector_type(8)))  float         v8f;
typedef __attribute__((ext_vector_type(4)))  unsigned int  v4u;
typedef __attribute__((ext_vector_type(8)))  int           v8i;
typedef __attribute__((ext_vector_type(4)))  int           v4i;

constexpr int Bn  = 32;
constexpr int Sn  = 256;
constexpr int Dn  = 512;
constexpr int Hn  = 8;
constexpr int DHn = 64;
constexpr int Fn  = 2048;
constexpr int Ln  = 6;
constexpr int Mn  = Bn * Sn;   // 8192 rows

union FragBF {
    v16bf v;
    uint4 u[2];
};

__device__ __forceinline__ unsigned short f2bf(float f) {
    unsigned u = __float_as_uint(f);
    unsigned r = u + 0x7FFFu + ((u >> 16) & 1u);   // round-to-nearest-even
    return (unsigned short)(r >> 16);
}

// Generic-pointer -> LDS byte offset (low 32 bits of the LDS-aperture address)
__device__ __forceinline__ unsigned lds_addr_of(const void* p) {
    return (unsigned)(unsigned long long)p;
}

#if __has_builtin(__builtin_amdgcn_tensor_load_to_lds) && \
    __has_builtin(__builtin_amdgcn_s_wait_tensorcnt)
#define USE_TDM 1
#else
#define USE_TDM 0
#endif

#if USE_TDM
// Post one TDM descriptor: 2-D tile (32 elems wide x tile_rows) of a row-major
// [rows_total x K_elems] bf16 tensor -> contiguous LDS tile [tile_rows][32].
// D# layout per CDNA5 ISA (data_size=2B, type=2).
__device__ __forceinline__ void tdm_load_tile(const unsigned short* gptr,
                                              unsigned lds_off,
                                              unsigned K_elems,
                                              unsigned rows_total,
                                              unsigned tile_rows)
{
    const unsigned long long ga = (unsigned long long)gptr;   // byte address
    v4u g0;
    g0.x = 1u;                                                // count=1, user mode
    g0.y = lds_off;                                           // lds_addr (bytes)
    g0.z = (unsigned)ga;                                      // global_addr[31:0]
    g0.w = (unsigned)((ga >> 32) & 0x01FFFFFFu) | 0x80000000u; // addr[56:32] | type=2

    v8i g1;
    g1[0] = (int)(1u << 16);                                  // data_size=1 (2 bytes)
    g1[1] = (int)((K_elems & 0xFFFFu) << 16);                 // tensor_dim0[15:0]
    g1[2] = (int)((K_elems >> 16) | ((rows_total & 0xFFFFu) << 16)); // dim0 hi | dim1 lo
    g1[3] = (int)((rows_total >> 16) | (32u << 16));          // dim1 hi | tile_dim0=32
    g1[4] = (int)(tile_rows & 0xFFFFu);                       // tile_dim1 (tile_dim2=0)
    g1[5] = (int)K_elems;                                     // tensor_dim0_stride[31:0]
    g1[6] = 0;                                                // stride hi / dim1_stride lo
    g1[7] = 0;

    const v4i gz = {0, 0, 0, 0};
#if defined(__clang_major__) && (__clang_major__ >= 23)
    const v8i gz8 = {0, 0, 0, 0, 0, 0, 0, 0};
    __builtin_amdgcn_tensor_load_to_lds(g0, g1, gz, gz, gz8, 0);
#else
    __builtin_amdgcn_tensor_load_to_lds(g0, g1, gz, gz, 0);
#endif
}
#endif // USE_TDM

// ---------------------------------------------------------------------------
// Generic bf16 GEMM: C[M x N] = act(A[M x K] * W[K x N] + bias[N]),  M == Mn.
// W supplied transposed (Bt[N x K], row-major).
// Block: 256 threads (8 waves); block tile 64x128; per-wave 32x32 (2x2 WMMA).
// TDM path: double-buffered LDS tiles (A 64x32, B 128x32) filled by the
// Tensor Data Mover (wave 0 posts descriptors), fragments read via ds_load.
// ---------------------------------------------------------------------------
__global__ __launch_bounds__(256) void gemm_bf16_kernel(
    const unsigned short* __restrict__ A,
    const unsigned short* __restrict__ Bt,
    const float* __restrict__ bias,
    float* __restrict__ Cf,
    unsigned short* __restrict__ Cb,
    int N, int K, int relu)
{
    const int tid  = threadIdx.x;
    const int wave = tid >> 5;
    const int lane = tid & 31;
    const int m    = lane & 15;
    const int hi   = lane >> 4;

    const int row0b = blockIdx.y * 64;
    const int col0b = blockIdx.x * 128;
    const int wrow  = (wave >> 2) * 32;
    const int wcol  = (wave & 3) * 32;
    const int row0  = row0b + wrow;
    const int col0  = col0b + wcol;

    v8f acc[2][2] = {};

#if USE_TDM
    __shared__ __attribute__((aligned(16))) unsigned short Atile[2][64][32];
    __shared__ __attribute__((aligned(16))) unsigned short Btile[2][128][32];

    const int nk = K >> 5;
    if (wave == 0) {   // EXEC-independent DMA; one wave owns the descriptors
        tdm_load_tile(A  + (size_t)row0b * K, lds_addr_of(&Atile[0][0][0]), K, (unsigned)Mn, 64);
        tdm_load_tile(Bt + (size_t)col0b * K, lds_addr_of(&Btile[0][0][0]), K, (unsigned)N, 128);
    }
    for (int ks = 0; ks < nk; ++ks) {
        const int cur = ks & 1;
        if (wave == 0) __builtin_amdgcn_s_wait_tensorcnt(0);  // cur tiles landed
        __syncthreads();                                      // visible to all waves
        if (wave == 0 && ks + 1 < nk) {                       // overlap next DMA
            const int k2 = (ks + 1) << 5;
            tdm_load_tile(A  + (size_t)row0b * K + k2,
                          lds_addr_of(&Atile[cur ^ 1][0][0]), K, (unsigned)Mn, 64);
            tdm_load_tile(Bt + (size_t)col0b * K + k2,
                          lds_addr_of(&Btile[cur ^ 1][0][0]), K, (unsigned)N, 128);
        }
        FragBF a[2], b[2];
#pragma unroll
        for (int i = 0; i < 2; ++i) {
            const unsigned short* ap = &Atile[cur][wrow + i * 16 + m][hi * 8];
            a[i].u[0] = *(const uint4*)(ap);
            a[i].u[1] = *(const uint4*)(ap + 16);
        }
#pragma unroll
        for (int j = 0; j < 2; ++j) {
            const unsigned short* bp = &Btile[cur][wcol + j * 16 + m][hi * 8];
            b[j].u[0] = *(const uint4*)(bp);
            b[j].u[1] = *(const uint4*)(bp + 16);
        }
#pragma unroll
        for (int i = 0; i < 2; ++i)
#pragma unroll
            for (int j = 0; j < 2; ++j)
                acc[i][j] = __builtin_amdgcn_wmma_f32_16x16x32_bf16(
                    false, a[i].v, false, b[j].v, (short)0, acc[i][j], false, false);
        __syncthreads();   // all reads of buf[cur] done before it is re-filled
    }
#else
    for (int k = 0; k < K; k += 32) {
        FragBF a[2], b[2];
        const unsigned short* ap0 = A + (size_t)(row0 + m) * K + k + hi * 8;
        const unsigned short* ap1 = ap0 + (size_t)16 * K;
        a[0].u[0] = *(const uint4*)(ap0);
        a[0].u[1] = *(const uint4*)(ap0 + 16);
        a[1].u[0] = *(const uint4*)(ap1);
        a[1].u[1] = *(const uint4*)(ap1 + 16);
        const unsigned short* bp0 = Bt + (size_t)(col0 + m) * K + k + hi * 8;
        const unsigned short* bp1 = bp0 + (size_t)16 * K;
        b[0].u[0] = *(const uint4*)(bp0);
        b[0].u[1] = *(const uint4*)(bp0 + 16);
        b[1].u[0] = *(const uint4*)(bp1);
        b[1].u[1] = *(const uint4*)(bp1 + 16);
#pragma unroll
        for (int i = 0; i < 2; ++i)
#pragma unroll
            for (int j = 0; j < 2; ++j)
                acc[i][j] = __builtin_amdgcn_wmma_f32_16x16x32_bf16(
                    false, a[i].v, false, b[j].v, (short)0, acc[i][j], false, false);
    }
#endif

#pragma unroll
    for (int i = 0; i < 2; ++i) {
#pragma unroll
        for (int j = 0; j < 2; ++j) {
            const int colg = col0 + j * 16 + m;
            const float bv = bias[colg];
#pragma unroll
            for (int r = 0; r < 8; ++r) {
                float v = acc[i][j][r] + bv;
                if (relu) v = fmaxf(v, 0.f);
                const size_t o = (size_t)(row0 + i * 16 + 8 * hi + r) * N + colg;
                if (Cf) Cf[o] = v;
                if (Cb) Cb[o] = f2bf(v);
            }
        }
    }
}

// ---------------------------------------------------------------------------
// Fused AKT attention for one (batch, head, 16-row query block).
// Phase 1: scores (Q.K^T / 8) via WMMA -> LDS (fp32)
// Phase 2: masked softmax -> cumsum -> distance decay -> masked softmax (fp32)
// Phase 3: attn @ V via WMMA reading attn from LDS, V pre-transposed
// ---------------------------------------------------------------------------
__global__ __launch_bounds__(256) void akt_attn_kernel(
    const unsigned short* __restrict__ Q,
    const unsigned short* __restrict__ Kt,
    const unsigned short* __restrict__ Vt,    // [B][H][DH][S]
    const float* __restrict__ gam,            // [H]
    unsigned short* __restrict__ Out,         // [B][S][D] bf16
    int offs, int zero_pad)
{
    __shared__ __attribute__((aligned(16))) float          sc[16][Sn];
    __shared__ __attribute__((aligned(16))) unsigned short at[16][Sn];

    const int tid  = threadIdx.x;
    const int wave = tid >> 5;
    const int lane = tid & 31;
    const int m    = lane & 15;
    const int hi   = lane >> 4;

    const int qb = blockIdx.x & 15;
    const int bh = blockIdx.x >> 4;
    const int h  = bh & (Hn - 1);
    const int b  = bh >> 3;
    const int q0 = qb * 16;

    const unsigned short* Qb = Q  + ((size_t)b * Sn + q0) * Dn + h * DHn;
    const unsigned short* Kb = Kt + (size_t)b * Sn * Dn + h * DHn;

    // ---- Phase 1: scores -> LDS -------------------------------------------
#pragma unroll
    for (int t = 0; t < 2; ++t) {
        const int key0 = (wave * 2 + t) * 16;
        v8f acc = {};
#pragma unroll
        for (int k = 0; k < DHn; k += 32) {
            FragBF a, bb;
            const unsigned short* ap = Qb + (size_t)m * Dn + k + hi * 8;
            a.u[0] = *(const uint4*)(ap);
            a.u[1] = *(const uint4*)(ap + 16);
            const unsigned short* bp = Kb + (size_t)(key0 + m) * Dn + k + hi * 8;
            bb.u[0] = *(const uint4*)(bp);
            bb.u[1] = *(const uint4*)(bp + 16);
            acc = __builtin_amdgcn_wmma_f32_16x16x32_bf16(
                false, a.v, false, bb.v, (short)0, acc, false, false);
        }
#pragma unroll
        for (int r = 0; r < 8; ++r)
            sc[8 * hi + r][key0 + m] = acc[r] * 0.125f;   // 1/sqrt(64)
    }
    __syncthreads();

    // ---- Phase 2: AKT row math (16 threads per query row) -----------------
    {
        const int r  = tid >> 4;
        const int c  = tid & 15;
        const int qr = q0 + r;
        const int limit = qr + offs - 1;   // max attendable key (inclusive)

        float s[16];
#pragma unroll
        for (int j = 0; j < 16; ++j) s[j] = sc[r][c * 16 + j];

        float m1 = -3.0e38f;
#pragma unroll
        for (int j = 0; j < 16; ++j)
            if (c * 16 + j <= limit) m1 = fmaxf(m1, s[j]);
        for (int d = 8; d > 0; d >>= 1) m1 = fmaxf(m1, __shfl_xor(m1, d, 16));

        float e[16];
        float ls = 0.f;
#pragma unroll
        for (int j = 0; j < 16; ++j) {
            e[j] = (c * 16 + j <= limit) ? __expf(s[j] - m1) : 0.f;
            ls += e[j];
        }
        float tot = ls;
        for (int d = 8; d > 0; d >>= 1) tot += __shfl_xor(tot, d, 16);
        const float inv = (tot > 0.f) ? 1.f / tot : 0.f;

        // inclusive scan across threads of this row (width-16 shfl)
        const float lsum = ls * inv;
        float scan = lsum;
        for (int d = 1; d < 16; d <<= 1) {
            float v = __shfl_up(scan, d, 16);
            if (c >= d) scan += v;
        }
        float run = scan - lsum;                 // exclusive prefix for this thread

        const float g     = gam[h];
        const float gamma = -log1pf(__expf(g));  // -softplus
        const float dtot  = (tot > 0.f) ? 1.f : 0.f;

        float s2[16];
#pragma unroll
        for (int j = 0; j < 16; ++j) {
            const float p = e[j] * inv;
            run += p;                             // inclusive cumsum
            const float pos  = fabsf((float)(qr - (c * 16 + j)));
            const float dist = sqrtf(fmaxf((dtot - run) * pos, 0.f));
            float te = __expf(dist * gamma);
            te = fminf(fmaxf(te, 1e-5f), 1e5f);
            s2[j] = (c * 16 + j <= limit) ? s[j] * te : -3.0e38f;
        }

        float m2 = -3.0e38f;
#pragma unroll
        for (int j = 0; j < 16; ++j) m2 = fmaxf(m2, s2[j]);
        for (int d = 8; d > 0; d >>= 1) m2 = fmaxf(m2, __shfl_xor(m2, d, 16));

        float e2[16];
        float ls2 = 0.f;
#pragma unroll
        for (int j = 0; j < 16; ++j) {
            e2[j] = (c * 16 + j <= limit) ? __expf(s2[j] - m2) : 0.f;
            ls2 += e2[j];
        }
        float tot2 = ls2;
        for (int d = 8; d > 0; d >>= 1) tot2 += __shfl_xor(tot2, d, 16);
        float inv2 = (tot2 > 0.f) ? 1.f / tot2 : 0.f;
        if (zero_pad && qr == 0) inv2 = 0.f;      // zero-pad first query row

#pragma unroll
        for (int j = 0; j < 16; ++j) at[r][c * 16 + j] = f2bf(e2[j] * inv2);
    }
    __syncthreads();

    // ---- Phase 3: attn(16x256) @ V(256x64) --------------------------------
    if (wave < 4) {
        const int d0 = wave * 16;
        const unsigned short* Vb =
            Vt + (((size_t)b * Hn + h) * DHn + d0 + m) * Sn;
        v8f acc = {};
        for (int k = 0; k < Sn; k += 32) {
            FragBF a, bb;
            const unsigned short* ap = &at[m][k + hi * 8];       // LDS reads
            a.u[0] = *(const uint4*)(ap);
            a.u[1] = *(const uint4*)(ap + 16);
            const unsigned short* bp = Vb + k + hi * 8;
            bb.u[0] = *(const uint4*)(bp);
            bb.u[1] = *(const uint4*)(bp + 16);
            acc = __builtin_amdgcn_wmma_f32_16x16x32_bf16(
                false, a.v, false, bb.v, (short)0, acc, false, false);
        }
#pragma unroll
        for (int r = 0; r < 8; ++r)
            Out[((size_t)b * Sn + q0 + 8 * hi + r) * Dn + h * DHn + d0 + m] =
                f2bf(acc[r]);
    }
}

// ---------------------------------------------------------------------------
// Fused residual-add + LayerNorm (D=512). One block per row, 256 threads.
// ---------------------------------------------------------------------------
__global__ __launch_bounds__(256) void add_ln_kernel(
    const float* __restrict__ resid, const float* __restrict__ delta,
    const float* __restrict__ g, const float* __restrict__ be,
    float* __restrict__ outF, unsigned short* __restrict__ outB)
{
    __shared__ float red[256];
    const int row = blockIdx.x;
    const int tid = threadIdx.x;
    const float* rp = resid + (size_t)row * Dn;
    const float* dp = delta + (size_t)row * Dn;

    const float v0 = rp[tid]       + dp[tid];
    const float v1 = rp[tid + 256] + dp[tid + 256];

    red[tid] = v0 + v1;
    __syncthreads();
    for (int st = 128; st > 0; st >>= 1) {
        if (tid < st) red[tid] += red[tid + st];
        __syncthreads();
    }
    const float mean = red[0] * (1.f / Dn);
    __syncthreads();

    const float a0 = v0 - mean, a1 = v1 - mean;
    red[tid] = a0 * a0 + a1 * a1;
    __syncthreads();
    for (int st = 128; st > 0; st >>= 1) {
        if (tid < st) red[tid] += red[tid + st];
        __syncthreads();
    }
    const float rstd = rsqrtf(red[0] * (1.f / Dn) + 1e-5f);

    const float o0 = a0 * rstd * g[tid]       + be[tid];
    const float o1 = a1 * rstd * g[tid + 256] + be[tid + 256];
    const size_t o = (size_t)row * Dn;
    outF[o + tid]       = o0;
    outF[o + tid + 256] = o1;
    if (outB) {
        outB[o + tid]       = f2bf(o0);
        outB[o + tid + 256] = f2bf(o1);
    }
}

// ---------------------------------------------------------------------------
// Helpers: weight transpose+pack, V transpose, activation copy+pack
// ---------------------------------------------------------------------------
__global__ void wprep_kernel(const float* __restrict__ W,
                             unsigned short* __restrict__ Wt, int K, int N)
{
    const int idx = blockIdx.x * 256 + threadIdx.x;   // over K*N
    const int n = idx % N;
    const int k = idx / N;
    Wt[(size_t)n * K + k] = f2bf(W[idx]);
}

__global__ void transpose_v_kernel(const unsigned short* __restrict__ V,
                                   unsigned short* __restrict__ Vt)
{
    const int idx = blockIdx.x * 256 + threadIdx.x;   // over B*H*DH*S
    const int s  = idx & (Sn - 1);
    int rest     = idx >> 8;
    const int d  = rest & (DHn - 1);
    rest >>= 6;
    const int h  = rest & (Hn - 1);
    const int bb = rest >> 3;
    Vt[idx] = V[((size_t)bb * Sn + s) * Dn + h * DHn + d];
}

__global__ void copy_pack_kernel(const float* __restrict__ src,
                                 float* __restrict__ dF,
                                 unsigned short* __restrict__ dB, int n)
{
    const int i = blockIdx.x * 256 + threadIdx.x;
    if (i < n) {
        const float v = src[i];
        dF[i] = v;
        dB[i] = f2bf(v);
    }
}

// ---------------------------------------------------------------------------
// Orchestration
// ---------------------------------------------------------------------------
extern "C" void kernel_launch(void* const* d_in, const int* in_sizes, int n_in,
                              void* d_out, int out_size, void* d_ws, size_t ws_size,
                              hipStream_t stream)
{
    const float* qemb = (const float*)d_in[0];
    const float* iemb = (const float*)d_in[1];
    const float* Wk   = (const float*)d_in[3];
    const float* bk   = (const float*)d_in[4];
    const float* Wv   = (const float*)d_in[5];
    const float* bv   = (const float*)d_in[6];
    const float* Wo   = (const float*)d_in[7];
    const float* bo   = (const float*)d_in[8];
    const float* gam  = (const float*)d_in[9];
    const float* ln1g = (const float*)d_in[10];
    const float* ln1b = (const float*)d_in[11];
    const float* W1   = (const float*)d_in[12];
    const float* b1   = (const float*)d_in[13];
    const float* W2   = (const float*)d_in[14];
    const float* b2   = (const float*)d_in[15];
    const float* ln2g = (const float*)d_in[16];
    const float* ln2b = (const float*)d_in[17];

    char* base = (char*)d_ws;
    size_t off = 0;
    auto alloc = [&](size_t bytes) -> void* {
        void* p = base + off;
        off += (bytes + 255) & ~(size_t)255;
        return p;
    };

    const size_t MDf = (size_t)Mn * Dn * sizeof(float);
    const size_t MDh = (size_t)Mn * Dn * sizeof(unsigned short);
    const size_t MFh = (size_t)Mn * Fn * sizeof(unsigned short);

    float*          yF = (float*)alloc(MDf);
    unsigned short* yB = (unsigned short*)alloc(MDh);
    float*          xF = (float*)alloc(MDf);
    unsigned short* xB = (unsigned short*)alloc(MDh);
    float*          tF = (float*)alloc(MDf);
    unsigned short* tB = (unsigned short*)alloc(MDh);
    float*          oF = (float*)alloc(MDf);
    unsigned short* qB = (unsigned short*)alloc(MDh);
    unsigned short* kB = (unsigned short*)alloc(MDh);
    unsigned short* vB = (unsigned short*)alloc(MDh);
    unsigned short* vT = (unsigned short*)alloc(MDh);
    unsigned short* aB = (unsigned short*)alloc(MDh);
    unsigned short* f1 = (unsigned short*)alloc(MFh);
    unsigned short* wkt = (unsigned short*)alloc((size_t)Ln * Dn * Dn * 2);
    unsigned short* wvt = (unsigned short*)alloc((size_t)Ln * Dn * Dn * 2);
    unsigned short* wot = (unsigned short*)alloc((size_t)Ln * Dn * Dn * 2);
    unsigned short* w1t = (unsigned short*)alloc((size_t)Ln * Dn * Fn * 2);
    unsigned short* w2t = (unsigned short*)alloc((size_t)Ln * Fn * Dn * 2);

    // ---- weight prep (deterministic, recomputed each call) ----------------
    const int nDD = Dn * Dn, nDF = Dn * Fn;
    for (int i = 0; i < Ln; ++i) {
        wprep_kernel<<<nDD / 256, 256, 0, stream>>>(Wk + (size_t)i * nDD, wkt + (size_t)i * nDD, Dn, Dn);
        wprep_kernel<<<nDD / 256, 256, 0, stream>>>(Wv + (size_t)i * nDD, wvt + (size_t)i * nDD, Dn, Dn);
        wprep_kernel<<<nDD / 256, 256, 0, stream>>>(Wo + (size_t)i * nDD, wot + (size_t)i * nDD, Dn, Dn);
        wprep_kernel<<<nDF / 256, 256, 0, stream>>>(W1 + (size_t)i * nDF, w1t + (size_t)i * nDF, Dn, Fn);
        wprep_kernel<<<nDF / 256, 256, 0, stream>>>(W2 + (size_t)i * nDF, w2t + (size_t)i * nDF, Fn, Dn);
    }

    const int nMD = Mn * Dn;
    copy_pack_kernel<<<nMD / 256, 256, 0, stream>>>(qemb, xF, xB, nMD);
    copy_pack_kernel<<<nMD / 256, 256, 0, stream>>>(iemb, yF, yB, nMD);

    const dim3 gD(Dn / 128, Mn / 64);   // N=512 GEMMs
    const dim3 gF(Fn / 128, Mn / 64);   // N=2048 GEMM (FFN up)

    auto layer = [&](int i, float* xqF, unsigned short* xqBp, unsigned short* xkBp,
                     unsigned short* xvBp, float* outF, unsigned short* outB,
                     bool mask_flag, bool apply_pos) {
        const int offs = mask_flag ? 1 : 0;
        const int zp   = mask_flag ? 0 : 1;
        const size_t oDD = (size_t)i * Dn * Dn;
        const size_t oD  = (size_t)i * Dn;
        const size_t oDF = (size_t)i * Dn * Fn;
        const size_t oFv = (size_t)i * Fn;

        // projections (query uses key weights: key_query_same=True)
        gemm_bf16_kernel<<<gD, 256, 0, stream>>>(xqBp, wkt + oDD, bk + oD, nullptr, qB, Dn, Dn, 0);
        gemm_bf16_kernel<<<gD, 256, 0, stream>>>(xkBp, wkt + oDD, bk + oD, nullptr, kB, Dn, Dn, 0);
        gemm_bf16_kernel<<<gD, 256, 0, stream>>>(xvBp, wvt + oDD, bv + oD, nullptr, vB, Dn, Dn, 0);
        transpose_v_kernel<<<nMD / 256, 256, 0, stream>>>(vB, vT);

        akt_attn_kernel<<<Bn * Hn * (Sn / 16), 256, 0, stream>>>(
            qB, kB, vT, gam + (size_t)i * Hn, aB, offs, zp);

        gemm_bf16_kernel<<<gD, 256, 0, stream>>>(aB, wot + oDD, bo + oD, oF, nullptr, Dn, Dn, 0);

        if (!apply_pos) {
            add_ln_kernel<<<Mn, 256, 0, stream>>>(xqF, oF, ln1g + oD, ln1b + oD, outF, outB);
        } else {
            add_ln_kernel<<<Mn, 256, 0, stream>>>(xqF, oF, ln1g + oD, ln1b + oD, tF, tB);
            gemm_bf16_kernel<<<gF, 256, 0, stream>>>(tB, w1t + oDF, b1 + oFv, nullptr, f1, Fn, Dn, 1);
            gemm_bf16_kernel<<<gD, 256, 0, stream>>>(f1, w2t + oDF, b2 + oD, oF, nullptr, Dn, Fn, 0);
            add_ln_kernel<<<Mn, 256, 0, stream>>>(tF, oF, ln2g + oD, ln2b + oD, outF, outB);
        }
    };

    // knowledge encoder: y = layer(y,y,y) x2, mask_flag=True, apply_pos=True
    layer(0, yF, yB, yB, yB, yF, yB, true, true);
    layer(1, yF, yB, yB, yB, yF, yB, true, true);
    // question encoder, alternating blocks
    layer(2, xF, xB, xB, xB, xF, xB, true,  false);  // self, no FFN
    layer(3, xF, xB, xB, yB, xF, xB, false, true);   // cross to y, zero_pad
    layer(4, xF, xB, xB, xB, xF, xB, true,  false);
    layer(5, xF, xB, xB, yB, (float*)d_out, nullptr, false, true);  // final -> d_out
}
